// TransformerEncoderLayer_6090263625988
// MI455X (gfx1250) — matrix operand order repaired and checked by reading, not compiled, not run on hardware
//
#include <hip/hip_runtime.h>

// ---------------------------------------------------------------------------
// Types for CDNA5 WMMA
// ---------------------------------------------------------------------------
typedef __attribute__((ext_vector_type(16))) __bf16 v16bf;
typedef __attribute__((ext_vector_type(8)))  float  v8f;

__device__ __forceinline__ unsigned short f32_to_bf16(float f) {
  union { float f; unsigned int u; } x; x.f = f;
  unsigned int r = x.u + 0x7FFFu + ((x.u >> 16) & 1u);
  return (unsigned short)(r >> 16);
}

// ---------------------------------------------------------------------------
// bf16 GEMM:  C[M,N] = scale * (A[M,K] @ B[N,K]^T) + bias[N]  (B stored N-major)
// Optional ReLU, optional f32 and/or bf16 output.
// Block: 256 threads (8 waves). Tile: 128x128, K-step 32 (one WMMA).
// Staging: double-buffered GLOBAL_LOAD_ASYNC_TO_LDS_B128 (ASYNCcnt pipeline).
// ---------------------------------------------------------------------------
#define BM 128
#define BN 128
#define BKS 32
#define LSTR 40   // ushorts per LDS row (32 data + 8 pad -> 80B stride)

__global__ void __launch_bounds__(256)
gemm_bf16_wmma(const unsigned short* __restrict__ A, int lda,
               const unsigned short* __restrict__ B, int ldb,
               float* __restrict__ Cf, unsigned short* __restrict__ Cb, int ldc,
               const float* __restrict__ bias, float scale, int relu,
               int M, int N, int K)
{
  __shared__ uint4 sAq[2][BM * LSTR / 8];
  __shared__ uint4 sBq[2][BN * LSTR / 8];

  const int tid   = threadIdx.x;
  const int wave  = tid >> 5;
  const int lane  = tid & 31;
  const int waveM = wave >> 1;            // 0..3 -> 32 rows each
  const int waveN = wave & 1;             // 0..1 -> 64 cols each
  const int mBase = waveM * 32;
  const int nBase = waveN * 64;
  const int lhalf = lane >> 4;            // 0/1
  const int l16   = lane & 15;

  v8f acc[2][4];
  #pragma unroll
  for (int i = 0; i < 2; ++i)
    #pragma unroll
    for (int j = 0; j < 4; ++j)
      acc[i][j] = {};

  const int blockM = blockIdx.y * BM;
  const int blockN = blockIdx.x * BN;

  // Issue async DMA of one 128x32 A tile + one 128x32 B tile into buffer `buf`.
  // 512 16B chunks per tile, 2 per thread per tile.
  auto stage = [&](int buf, int k0) {
    #pragma unroll
    for (int c = 0; c < 2; ++c) {
      int chunk = tid + c * 256;          // 0..511
      int row = chunk >> 2;
      int seg = chunk & 3;
      int gm = blockM + row; if (gm >= M) gm = M - 1;
      int gn = blockN + row; if (gn >= N) gn = N - 1;
      unsigned la = (unsigned)(uintptr_t)
          ((unsigned short*)sAq[buf] + row * LSTR + seg * 8);
      unsigned long long ga = (unsigned long long)(uintptr_t)
          (A + (size_t)gm * lda + k0 + seg * 8);
      asm volatile("global_load_async_to_lds_b128 %0, %1, off"
                   :: "v"(la), "v"(ga) : "memory");
      unsigned lb = (unsigned)(uintptr_t)
          ((unsigned short*)sBq[buf] + row * LSTR + seg * 8);
      unsigned long long gb = (unsigned long long)(uintptr_t)
          (B + (size_t)gn * ldb + k0 + seg * 8);
      asm volatile("global_load_async_to_lds_b128 %0, %1, off"
                   :: "v"(lb), "v"(gb) : "memory");
    }
  };

  stage(0, 0);
  asm volatile("s_wait_asynccnt 0x0" ::: "memory");
  __syncthreads();

  int buf = 0;
  for (int k0 = 0; k0 < K; k0 += BKS) {
    if (k0 + BKS < K) stage(buf ^ 1, k0 + BKS);   // prefetch next tile via DMA

    const unsigned short* sA = (const unsigned short*)sAq[buf];
    const unsigned short* sB = (const unsigned short*)sBq[buf];

    union Frag { uint4 u[2]; v16bf v; };
    Frag af[2]; Frag bfr[4];
    // A 16x32 fragment: lanes 0-15 hold K 0-7 / 16-23, lanes 16-31 hold K 8-15 / 24-31
    #pragma unroll
    for (int m = 0; m < 2; ++m) {
      int row = mBase + m * 16 + l16;
      af[m].u[0] = *(const uint4*)(sA + row * LSTR + lhalf * 8);
      af[m].u[1] = *(const uint4*)(sA + row * LSTR + lhalf * 8 + 16);
    }
    // B 32x16 fragment: lane = column n; lanes 0-15 hold K 0-7 / 16-23, lanes 16-31 hold K 8-15 / 24-31
    #pragma unroll
    for (int n = 0; n < 4; ++n) {
      int col = nBase + n * 16 + l16;
      bfr[n].u[0] = *(const uint4*)(sB + col * LSTR + lhalf * 8);
      bfr[n].u[1] = *(const uint4*)(sB + col * LSTR + lhalf * 8 + 16);
    }
    #pragma unroll
    for (int m = 0; m < 2; ++m)
      #pragma unroll
      for (int n = 0; n < 4; ++n)
        acc[m][n] = __builtin_amdgcn_wmma_f32_16x16x32_bf16(
            false, af[m].v, false, bfr[n].v, (short)0, acc[m][n], false, false);

    // DMA for next tile must have landed before anyone reads it; current-buffer
    // reads are protected by the dscnt wait inside the barrier.
    asm volatile("s_wait_asynccnt 0x0" ::: "memory");
    __syncthreads();
    buf ^= 1;
  }

  // Epilogue. C/D layout: VGPR r -> row r (+8 for lanes 16-31), col = lane&15.
  #pragma unroll
  for (int m = 0; m < 2; ++m) {
    int gmBase = blockM + mBase + m * 16 + lhalf * 8;
    #pragma unroll
    for (int n = 0; n < 4; ++n) {
      int gn = blockN + nBase + n * 16 + l16;
      if (gn >= N) continue;
      float bval = bias ? bias[gn] : 0.0f;
      #pragma unroll
      for (int r = 0; r < 8; ++r) {
        int gm = gmBase + r;
        if (gm >= M) continue;
        float v = acc[m][n][r] * scale + bval;
        if (relu) v = fmaxf(v, 0.0f);
        if (Cf) Cf[(size_t)gm * ldc + gn] = v;
        if (Cb) Cb[(size_t)gm * ldc + gn] = f32_to_bf16(v);
      }
    }
  }
}

// ---------------------------------------------------------------------------
// Row softmax (f32 in, bf16 out). One 256-thread block per row.
// ---------------------------------------------------------------------------
__global__ void __launch_bounds__(256)
softmax_row(const float* __restrict__ x, unsigned short* __restrict__ y, int n)
{
  __shared__ float red[256];
  const int row = blockIdx.x, tid = threadIdx.x;
  const float* xr = x + (size_t)row * n;
  float m = -3.0e38f;
  for (int i = tid; i < n; i += 256) m = fmaxf(m, xr[i]);
  red[tid] = m; __syncthreads();
  for (int s = 128; s > 0; s >>= 1) {
    if (tid < s) red[tid] = fmaxf(red[tid], red[tid + s]);
    __syncthreads();
  }
  m = red[0]; __syncthreads();
  float sum = 0.f;
  for (int i = tid; i < n; i += 256) sum += __expf(xr[i] - m);
  red[tid] = sum; __syncthreads();
  for (int s = 128; s > 0; s >>= 1) {
    if (tid < s) red[tid] += red[tid + s];
    __syncthreads();
  }
  float inv = 1.0f / red[0];
  for (int i = tid; i < n; i += 256)
    y[(size_t)row * n + i] = f32_to_bf16(__expf(xr[i] - m) * inv);
}

// ---------------------------------------------------------------------------
// out = LayerNorm(a + b) * g + be  (f32, optional bf16 copy). One block/row.
// ---------------------------------------------------------------------------
__global__ void __launch_bounds__(256)
add_layernorm(const float* __restrict__ a, const float* __restrict__ b,
              const float* __restrict__ g, const float* __restrict__ be,
              float* __restrict__ outf, unsigned short* __restrict__ outb, int D_)
{
  __shared__ float red[256];
  const int row = blockIdx.x, tid = threadIdx.x;
  const int per = D_ / 256;          // D=1024 -> 4
  float loc[8];
  float sum = 0.f;
  for (int r = 0; r < per; ++r) {
    int i = tid + r * 256;
    loc[r] = a[(size_t)row * D_ + i] + b[(size_t)row * D_ + i];
    sum += loc[r];
  }
  red[tid] = sum; __syncthreads();
  for (int s = 128; s > 0; s >>= 1) { if (tid < s) red[tid] += red[tid + s]; __syncthreads(); }
  float mu = red[0] / D_; __syncthreads();
  float vs = 0.f;
  for (int r = 0; r < per; ++r) { float d = loc[r] - mu; vs += d * d; }
  red[tid] = vs; __syncthreads();
  for (int s = 128; s > 0; s >>= 1) { if (tid < s) red[tid] += red[tid + s]; __syncthreads(); }
  float rstd = rsqrtf(red[0] / D_ + 1e-5f);
  for (int r = 0; r < per; ++r) {
    int i = tid + r * 256;
    float v = (loc[r] - mu) * rstd * g[i] + be[i];
    if (outf) outf[(size_t)row * D_ + i] = v;
    if (outb) outb[(size_t)row * D_ + i] = f32_to_bf16(v);
  }
}

// ---------------------------------------------------------------------------
// Packing / layout kernels
// ---------------------------------------------------------------------------
__global__ void cvt_f32_bf16(const float* __restrict__ in,
                             unsigned short* __restrict__ out, int n) {
  int i = blockIdx.x * 256 + threadIdx.x;
  if (i < n) out[i] = f32_to_bf16(in[i]);
}

// Wq/Wk/Wv (H,D,DK) f32 -> combined N-major bf16: out[n][d], n = which*HDK + h*DK + k
__global__ void pack_qkv_w(const float* __restrict__ Wq, const float* __restrict__ Wk,
                           const float* __restrict__ Wv, unsigned short* __restrict__ out,
                           int D_, int HDK_, int DK_, int total) {
  int idx = blockIdx.x * 256 + threadIdx.x;
  if (idx >= total) return;
  int n = idx / D_, d = idx % D_;
  int which = n / HDK_, r = n % HDK_;
  int h = r / DK_, k = r % DK_;
  const float* W = (which == 0) ? Wq : (which == 1) ? Wk : Wv;
  out[idx] = f32_to_bf16(W[(size_t)h * D_ * DK_ + (size_t)d * DK_ + k]);
}

__global__ void concat_bias(const float* __restrict__ bq, const float* __restrict__ bk,
                            const float* __restrict__ bv, float* __restrict__ out,
                            int HDK_, int total) {
  int idx = blockIdx.x * 256 + threadIdx.x;
  if (idx >= total) return;
  int which = idx / HDK_, r = idx % HDK_;
  const float* b = (which == 0) ? bq : (which == 1) ? bk : bv;
  out[idx] = b[r];
}

// in (K_,N_) f32 row-major -> out (N_,K_) bf16 row-major
__global__ void transpose_f32_bf16(const float* __restrict__ in,
                                   unsigned short* __restrict__ out,
                                   int K_, int N_, int total) {
  int idx = blockIdx.x * 256 + threadIdx.x;
  if (idx >= total) return;
  int n = idx / K_, k = idx % K_;
  out[idx] = f32_to_bf16(in[(size_t)k * N_ + n]);
}

// qkv (S, 3*HDK) bf16 -> vt (H*DK, S) bf16 (V section transposed)
__global__ void pack_vt(const unsigned short* __restrict__ qkv,
                        unsigned short* __restrict__ vt,
                        int S_, int ld, int voff, int total) {
  int idx = blockIdx.x * 256 + threadIdx.x;
  if (idx >= total) return;
  int t = idx % S_, r = idx / S_;
  vt[idx] = qkv[(size_t)t * ld + voff + r];
}

// ---------------------------------------------------------------------------
// Host orchestration
// ---------------------------------------------------------------------------
extern "C" void kernel_launch(void* const* d_in, const int* in_sizes, int n_in,
                              void* d_out, int out_size, void* d_ws, size_t ws_size,
                              hipStream_t stream)
{
  (void)in_sizes; (void)n_in; (void)out_size; (void)ws_size;
  const int S = 4096, D = 1024, H = 16, DK = 64, F = 4096;
  const int HDK = H * DK;        // 1024
  const int N3  = 3 * HDK;       // 3072

  const float* src  = (const float*)d_in[0];
  const float* Wq   = (const float*)d_in[1];
  const float* bq   = (const float*)d_in[2];
  const float* Wk   = (const float*)d_in[3];
  const float* bk   = (const float*)d_in[4];
  const float* Wv   = (const float*)d_in[5];
  const float* bv   = (const float*)d_in[6];
  const float* Wo   = (const float*)d_in[7];
  const float* bo   = (const float*)d_in[8];
  const float* g1   = (const float*)d_in[9];
  const float* be1  = (const float*)d_in[10];
  const float* W1   = (const float*)d_in[11];
  const float* b1   = (const float*)d_in[12];
  const float* W2   = (const float*)d_in[13];
  const float* b2   = (const float*)d_in[14];
  const float* g2   = (const float*)d_in[15];
  const float* be2  = (const float*)d_in[16];
  float* out = (float*)d_out;

  char* ws = (char*)d_ws;
  size_t off = 0;
  auto alloc = [&](size_t bytes) -> char* {
    char* p = ws + off;
    off = (off + bytes + 255) & ~(size_t)255;
    return p;
  };
  unsigned short* srcb   = (unsigned short*)alloc((size_t)S * D * 2);
  unsigned short* wqkvt  = (unsigned short*)alloc((size_t)N3 * D * 2);
  float*          bqkv   = (float*)         alloc((size_t)N3 * 4);
  unsigned short* wot    = (unsigned short*)alloc((size_t)D * HDK * 2);
  unsigned short* w1t    = (unsigned short*)alloc((size_t)F * D * 2);
  unsigned short* w2t    = (unsigned short*)alloc((size_t)D * F * 2);
  unsigned short* qkv    = (unsigned short*)alloc((size_t)S * N3 * 2);
  unsigned short* vt     = (unsigned short*)alloc((size_t)HDK * S * 2);
  float*          logits = (float*)         alloc((size_t)S * S * 4);
  unsigned short* attnb  = (unsigned short*)alloc((size_t)S * S * 2);
  unsigned short* conc   = (unsigned short*)alloc((size_t)S * HDK * 2);
  float*          attn_o = (float*)         alloc((size_t)S * D * 4);
  float*          x1f    = (float*)         alloc((size_t)S * D * 4);
  unsigned short* x1b    = (unsigned short*)alloc((size_t)S * D * 2);
  unsigned short* h1     = (unsigned short*)alloc((size_t)S * F * 2);
  float*          ffn    = (float*)         alloc((size_t)S * D * 4);

  auto cdiv = [](int a, int b) { return (a + b - 1) / b; };

  // ---- pack / convert ----
  { int n = S * D;   cvt_f32_bf16     <<<cdiv(n,256),256,0,stream>>>(src, srcb, n); }
  { int n = N3 * D;  pack_qkv_w       <<<cdiv(n,256),256,0,stream>>>(Wq, Wk, Wv, wqkvt, D, HDK, DK, n); }
  { int n = N3;      concat_bias      <<<cdiv(n,256),256,0,stream>>>(bq, bk, bv, bqkv, HDK, n); }
  { int n = D * HDK; transpose_f32_bf16<<<cdiv(n,256),256,0,stream>>>(Wo, wot, HDK, D, n); }
  { int n = F * D;   transpose_f32_bf16<<<cdiv(n,256),256,0,stream>>>(W1, w1t, D, F, n); }
  { int n = D * F;   transpose_f32_bf16<<<cdiv(n,256),256,0,stream>>>(W2, w2t, F, D, n); }

  // ---- fused QKV projection: [S,D] @ [D,3*HDK] + bias -> bf16 qkv ----
  gemm_bf16_wmma<<<dim3(N3/128, S/128), 256, 0, stream>>>(
      srcb, D, wqkvt, D, nullptr, qkv, N3, bqkv, 1.0f, 0, S, N3, D);

  // ---- V transpose to (H*DK, S) ----
  { int n = HDK * S; pack_vt<<<cdiv(n,256),256,0,stream>>>(qkv, vt, S, N3, 2*HDK, n); }

  // ---- per-head attention ----
  for (int h = 0; h < H; ++h) {
    const unsigned short* Qh = qkv + h * DK;          // [S,DK], ld = N3
    const unsigned short* Kh = qkv + HDK + h * DK;    // [S,DK], ld = N3 (B is N-major)
    gemm_bf16_wmma<<<dim3(S/128, S/128), 256, 0, stream>>>(
        Qh, N3, Kh, N3, logits, nullptr, S, nullptr, 0.125f, 0, S, S, DK);
    softmax_row<<<S, 256, 0, stream>>>(logits, attnb, S);
    gemm_bf16_wmma<<<dim3(1, S/128), 256, 0, stream>>>(
        attnb, S, vt + (size_t)h * DK * S, S,
        nullptr, conc + h * DK, HDK, nullptr, 1.0f, 0, S, DK, S);
  }

  // ---- output projection + residual LN ----
  gemm_bf16_wmma<<<dim3(D/128, S/128), 256, 0, stream>>>(
      conc, HDK, wot, HDK, attn_o, nullptr, D, bo, 1.0f, 0, S, D, HDK);
  add_layernorm<<<S, 256, 0, stream>>>(src, attn_o, g1, be1, x1f, x1b, D);

  // ---- FFN ----
  gemm_bf16_wmma<<<dim3(F/128, S/128), 256, 0, stream>>>(
      x1b, D, w1t, D, nullptr, h1, F, b1, 1.0f, 1, S, F, D);
  gemm_bf16_wmma<<<dim3(D/128, S/128), 256, 0, stream>>>(
      h1, F, w2t, F, ffn, nullptr, D, b2, 1.0f, 0, S, D, F);
  add_layernorm<<<S, 256, 0, stream>>>(x1f, ffn, g2, be2, out, nullptr, D);
}
